// GraphSAGE_77094662963915
// MI455X (gfx1250) — compile-verified, matrix operand order checked by reading, not run on hardware
//
#include <hip/hip_runtime.h>
#include <hip/hip_bf16.h>
#include <math.h>

// ---------------------------------------------------------------------------
// GraphSAGE (mean agg) 128 -> 256 -> 256 -> 40, BN+ReLU between layers,
// log_softmax output.  WMMA bf16 GEMMs (f32 accum), atomic scatter aggregation.
// Weights are pre-swizzled once per launch into bf16 WMMA-fragment layout so
// the GEMM inner loop issues only wide b128 loads (no guards, no column
// striding).
// ---------------------------------------------------------------------------

typedef __bf16 bf16_t;
typedef __attribute__((ext_vector_type(16))) __bf16 v16bf;
typedef __attribute__((ext_vector_type(8)))  float  v8f;

#define IN_F   128
#define HID_F  256
#define NCLS_F 40

// ---------------- degree ----------------------------------------------------
__global__ __launch_bounds__(256) void deg_count_kernel(const int* __restrict__ dst,
                                                        float* __restrict__ deg, int E) {
  int i = blockIdx.x * blockDim.x + threadIdx.x;
  if (i < E) atomicAdd(&deg[dst[i]], 1.0f);
}

__global__ __launch_bounds__(256) void make_invdeg_kernel(float* __restrict__ deg, int N) {
  int i = blockIdx.x * blockDim.x + threadIdx.x;
  if (i < N) {
    float d = deg[i];
    deg[i] = 1.0f / fmaxf(d, 1.0f);
  }
}

// ---------------- edge scatter-add (gather src row, atomic add into dst row)
// F is a power of two here (128 or 256); log2chunks = log2(F/4).
__global__ __launch_bounds__(256) void agg_scatter_kernel(const float* __restrict__ h,
                                                          const int* __restrict__ src,
                                                          const int* __restrict__ dst,
                                                          float* __restrict__ agg,
                                                          long long total, int F,
                                                          int log2chunks) {
  long long idx = (long long)blockIdx.x * blockDim.x + threadIdx.x;
  if (idx >= total) return;
  int e = (int)(idx >> log2chunks);
  int c = (int)(idx & ((1 << log2chunks) - 1));
  int s = src[e];
  int d = dst[e];
  const float4 v = *(const float4*)(h + (size_t)s * F + (size_t)c * 4);
  float* p = agg + (size_t)d * F + (size_t)c * 4;
  atomicAdd(p + 0, v.x);
  atomicAdd(p + 1, v.y);
  atomicAdd(p + 2, v.z);
  atomicAdd(p + 3, v.w);
}

// ---------------- weight pre-swizzle ----------------------------------------
// Source: W row-major [K, F] f32.  Dest: bf16 fragment-major, padded to Fp cols:
//   Wp[ ((kblk*Fp + n)*2 + half)*16 + i ]  holds W[k, n] with
//   k = kblk*32 + (i<8 ? half*8 + i : 16 + half*8 + (i-8))
// so a lane (col n, half = lane>>4) reads its whole 16-element B fragment as
// 32 contiguous, 32B-aligned bytes.
__global__ __launch_bounds__(256) void prep_w_kernel(const float* __restrict__ W,
                                                     bf16_t* __restrict__ Wp,
                                                     int K, int F, int Fp) {
  int idx = blockIdx.x * blockDim.x + threadIdx.x;
  int total = (K >> 5) * Fp * 32;
  if (idx >= total) return;
  int i    = idx & 15;
  int half = (idx >> 4) & 1;
  int n    = (idx >> 5) % Fp;
  int kblk = idx / (Fp * 32);
  int kb   = half * 8;
  int k    = kblk * 32 + (i < 8 ? kb + i : 16 + kb + (i - 8));
  Wp[idx] = (n < F) ? (bf16_t)W[(size_t)k * F + n] : (bf16_t)0.0f;
}

// ---------------- WMMA A-fragment (f32 -> bf16 in registers) ----------------
// A fragment 16x32 bf16: lane L holds row m = L&15; kb = (L<16)?0:8.
// elements 0..7 -> K = kb+i ; elements 8..15 -> K = 16+kb+(i-8)
__device__ __forceinline__ v16bf load_a_frag(const float* __restrict__ row, int kb, float scale) {
  v16bf a;
#pragma unroll
  for (int i = 0; i < 8; ++i) a[i] = (bf16_t)(row[kb + i] * scale);
#pragma unroll
  for (int i = 0; i < 8; ++i) a[8 + i] = (bf16_t)(row[16 + kb + i] * scale);
  return a;
}

// ---------------- fused dual GEMM: out = A1@W1 + (A2*invdeg)@W2 + bias ------
// block = 256 threads = 8 waves; wave w owns rows [blockIdx.x*128 + w*16, +16)
// and 64 output columns [blockIdx.y*64, +64) as four 16x16 WMMA tiles.
// W1p/W2p are pre-swizzled bf16 (see prep_w_kernel), padded to Fp columns.
__global__ __launch_bounds__(256)
void sage_gemm_kernel(const float* __restrict__ A1, const bf16_t* __restrict__ W1p,
                      const float* __restrict__ A2, const bf16_t* __restrict__ W2p,
                      const float* __restrict__ bias, const float* __restrict__ invdeg,
                      float* __restrict__ out, int M, int K, int F, int Fp) {
  const int lane    = threadIdx.x & 31;
  const int wave    = threadIdx.x >> 5;
  const int rowBase = blockIdx.x * 128 + wave * 16;
  if (rowBase >= M) return;                 // M % 16 == 0, so full tiles only
  const int colBase = blockIdx.y * 64;
  const int nlane   = lane & 15;
  const int half    = lane >> 4;            // 0 or 1
  const int kb      = half * 8;
  const int mRow    = rowBase + nlane;

  v8f acc[4];
#pragma unroll
  for (int t = 0; t < 4; ++t)
#pragma unroll
    for (int i = 0; i < 8; ++i) acc[t][i] = 0.0f;

  // per-lane B base: fragment elements for (col n, half) are 16 contiguous bf16
  const size_t bstep = (size_t)Fp * 32;       // elements per k-block (K += 32)
  const bf16_t* w1 = W1p + (size_t)(colBase + nlane) * 32 + (size_t)half * 16;
  const bf16_t* w2 = W2p + (size_t)(colBase + nlane) * 32 + (size_t)half * 16;

  // self term: A1 @ W1
  const float* a1row = A1 + (size_t)mRow * K;
  for (int k0 = 0; k0 < K; k0 += 32, w1 += bstep) {
    v16bf a = load_a_frag(a1row + k0, kb, 1.0f);
#pragma unroll
    for (int t = 0; t < 4; ++t) {
      v16bf b = *(const v16bf*)(w1 + t * 512);   // t*16 cols * 32 elem/col
      acc[t] = __builtin_amdgcn_wmma_f32_16x16x32_bf16(false, a, false, b,
                                                       (short)0, acc[t], false, false);
    }
  }

  // neighbor term: (A2 * invdeg[row]) @ W2  (deg-normalization fused into A load)
  const float  idg   = invdeg[mRow];
  const float* a2row = A2 + (size_t)mRow * K;
  for (int k0 = 0; k0 < K; k0 += 32, w2 += bstep) {
    v16bf a = load_a_frag(a2row + k0, kb, idg);
#pragma unroll
    for (int t = 0; t < 4; ++t) {
      v16bf b = *(const v16bf*)(w2 + t * 512);
      acc[t] = __builtin_amdgcn_wmma_f32_16x16x32_bf16(false, a, false, b,
                                                       (short)0, acc[t], false, false);
    }
  }

  // store: C/D layout -> element r of v8f at lane L is (m = r + half*8, n = L&15)
  const int rOff = half * 8;
#pragma unroll
  for (int t = 0; t < 4; ++t) {
    int n = colBase + t * 16 + nlane;
    if (n < F) {
      float bv = bias[n];
#pragma unroll
      for (int r = 0; r < 8; ++r)
        out[(size_t)(rowBase + rOff + r) * F + n] = acc[t][r] + bv;
    }
  }
}

// ---------------- BatchNorm (training-mode batch stats) ---------------------
// blockDim.x == F (256).  Each block reduces 128 rows, atomically accumulates.
__global__ void col_stats_kernel(const float* __restrict__ h, float* __restrict__ sum,
                                 float* __restrict__ sumsq, int M, int F) {
  int c = threadIdx.x;
  if (c >= F) return;
  int r0 = blockIdx.x * 128;
  int r1 = min(r0 + 128, M);
  float s = 0.0f, q = 0.0f;
  for (int r = r0; r < r1; ++r) {
    float v = h[(size_t)r * F + c];
    s += v;
    q += v * v;
  }
  atomicAdd(&sum[c], s);
  atomicAdd(&sumsq[c], q);
}

__global__ void bn_finalize_kernel(const float* __restrict__ sum, const float* __restrict__ sumsq,
                                   const float* __restrict__ g, const float* __restrict__ beta,
                                   float* __restrict__ scale, float* __restrict__ shift,
                                   int M, int F) {
  int c = blockIdx.x * blockDim.x + threadIdx.x;
  if (c >= F) return;
  float invM = 1.0f / (float)M;
  float mu   = sum[c] * invM;
  float var  = sumsq[c] * invM - mu * mu;
  float sc   = g[c] * rsqrtf(var + 1e-5f);
  scale[c] = sc;
  shift[c] = beta[c] - mu * sc;
}

__global__ __launch_bounds__(256) void bn_apply_relu_kernel(float* __restrict__ h,
                                                            const float* __restrict__ scale,
                                                            const float* __restrict__ shift,
                                                            long long total, int F) {
  long long i = (long long)blockIdx.x * blockDim.x + threadIdx.x;
  if (i >= total) return;
  int c = (int)(i & (F - 1));                 // F == 256 (power of two)
  float v = h[i] * scale[c] + shift[c];
  h[i] = fmaxf(v, 0.0f);
}

// ---------------- row-wise log_softmax (F = 40) ------------------------------
__global__ __launch_bounds__(256) void log_softmax_kernel(float* __restrict__ out, int M, int F) {
  int r = blockIdx.x * blockDim.x + threadIdx.x;
  if (r >= M) return;
  float* row = out + (size_t)r * F;
  float mx = -INFINITY;
  for (int j = 0; j < F; ++j) mx = fmaxf(mx, row[j]);
  float s = 0.0f;
  for (int j = 0; j < F; ++j) s += __expf(row[j] - mx);
  float lse = mx + __logf(s);
  for (int j = 0; j < F; ++j) row[j] -= lse;
}

// ---------------------------------------------------------------------------
extern "C" void kernel_launch(void* const* d_in, const int* in_sizes, int n_in,
                              void* d_out, int out_size, void* d_ws, size_t ws_size,
                              hipStream_t stream) {
  const float* x   = (const float*)d_in[0];
  const int*   src = (const int*)d_in[1];
  const int*   dst = (const int*)d_in[2];
  const float* Ws0 = (const float*)d_in[3];
  const float* Wn0 = (const float*)d_in[4];
  const float* b0  = (const float*)d_in[5];
  const float* Ws1 = (const float*)d_in[6];
  const float* Wn1 = (const float*)d_in[7];
  const float* b1  = (const float*)d_in[8];
  const float* Ws2 = (const float*)d_in[9];
  const float* Wn2 = (const float*)d_in[10];
  const float* b2  = (const float*)d_in[11];
  const float* g0  = (const float*)d_in[12];
  const float* be0 = (const float*)d_in[13];
  const float* g1  = (const float*)d_in[14];
  const float* be1 = (const float*)d_in[15];

  const int N = in_sizes[0] / IN_F;
  const int E = in_sizes[1];

  // workspace carve-up
  float* ws       = (float*)d_ws;
  float* invdeg   = ws;                             // N            (deg -> invdeg in place)
  float* h0       = invdeg + N;                     // N * 256
  float* h1       = h0 + (size_t)N * HID_F;         // N * 256
  float* agg      = h1 + (size_t)N * HID_F;         // N * 256
  float* bn_sum   = agg + (size_t)N * HID_F;        // 256
  float* bn_sq    = bn_sum + HID_F;                 // 256
  float* bn_scale = bn_sq + HID_F;                  // 256
  float* bn_shift = bn_scale + HID_F;               // 256
  size_t foff = (size_t)N + 3 * (size_t)N * HID_F + 4 * HID_F;
  foff = (foff + 15) & ~(size_t)15;                 // 64B-align the bf16 region
  bf16_t* wp      = (bf16_t*)(ws + foff);
  const int FP_HID = HID_F;                         // 256 (already mult of 64)
  const int FP_CLS = 64;                            // 40 padded to 64
  bf16_t* wp_s0 = wp;                                       // (128/32)*256*32 = 32768
  bf16_t* wp_n0 = wp_s0 + (size_t)(IN_F  / 32) * FP_HID * 32;
  bf16_t* wp_s1 = wp_n0 + (size_t)(IN_F  / 32) * FP_HID * 32; // 65536
  bf16_t* wp_n1 = wp_s1 + (size_t)(HID_F / 32) * FP_HID * 32;
  bf16_t* wp_s2 = wp_n1 + (size_t)(HID_F / 32) * FP_HID * 32; // 16384
  bf16_t* wp_n2 = wp_s2 + (size_t)(HID_F / 32) * FP_CLS * 32;
  float* logits   = (float*)d_out;                  // N * 40 (log_softmax in place)

  // ---- weight pre-swizzle (tiny; runs while nothing else is queued) ----
  {
    int t0 = (IN_F  / 32) * FP_HID * 32;  // 32768
    int t1 = (HID_F / 32) * FP_HID * 32;  // 65536
    int t2 = (HID_F / 32) * FP_CLS * 32;  // 16384
    prep_w_kernel<<<(t0 + 255) / 256, 256, 0, stream>>>(Ws0, wp_s0, IN_F,  HID_F,  FP_HID);
    prep_w_kernel<<<(t0 + 255) / 256, 256, 0, stream>>>(Wn0, wp_n0, IN_F,  HID_F,  FP_HID);
    prep_w_kernel<<<(t1 + 255) / 256, 256, 0, stream>>>(Ws1, wp_s1, HID_F, HID_F,  FP_HID);
    prep_w_kernel<<<(t1 + 255) / 256, 256, 0, stream>>>(Wn1, wp_n1, HID_F, HID_F,  FP_HID);
    prep_w_kernel<<<(t2 + 255) / 256, 256, 0, stream>>>(Ws2, wp_s2, HID_F, NCLS_F, FP_CLS);
    prep_w_kernel<<<(t2 + 255) / 256, 256, 0, stream>>>(Wn2, wp_n2, HID_F, NCLS_F, FP_CLS);
  }

  // ---- degrees -> invdeg ----
  hipMemsetAsync(invdeg, 0, (size_t)N * sizeof(float), stream);
  deg_count_kernel<<<(E + 255) / 256, 256, 0, stream>>>(dst, invdeg, E);
  make_invdeg_kernel<<<(N + 255) / 256, 256, 0, stream>>>(invdeg, N);

  dim3 gemm_grid_hid((N + 127) / 128, (HID_F + 63) / 64);
  dim3 gemm_grid_cls((N + 127) / 128, 1);
  long long totHid = (long long)N * HID_F;

  // ---- layer 0: x (128) -> h0 (256) ----
  hipMemsetAsync(agg, 0, (size_t)N * IN_F * sizeof(float), stream);
  {
    long long tot = (long long)E * (IN_F / 4);
    agg_scatter_kernel<<<(int)((tot + 255) / 256), 256, 0, stream>>>(x, src, dst, agg, tot, IN_F, 5);
  }
  sage_gemm_kernel<<<gemm_grid_hid, 256, 0, stream>>>(x, wp_s0, agg, wp_n0, b0, invdeg, h0,
                                                      N, IN_F, HID_F, FP_HID);
  hipMemsetAsync(bn_sum, 0, 2 * HID_F * sizeof(float), stream);
  col_stats_kernel<<<(N + 127) / 128, HID_F, 0, stream>>>(h0, bn_sum, bn_sq, N, HID_F);
  bn_finalize_kernel<<<1, HID_F, 0, stream>>>(bn_sum, bn_sq, g0, be0, bn_scale, bn_shift, N, HID_F);
  bn_apply_relu_kernel<<<(int)((totHid + 255) / 256), 256, 0, stream>>>(h0, bn_scale, bn_shift, totHid, HID_F);

  // ---- layer 1: h0 (256) -> h1 (256) ----
  hipMemsetAsync(agg, 0, (size_t)N * HID_F * sizeof(float), stream);
  {
    long long tot = (long long)E * (HID_F / 4);
    agg_scatter_kernel<<<(int)((tot + 255) / 256), 256, 0, stream>>>(h0, src, dst, agg, tot, HID_F, 6);
  }
  sage_gemm_kernel<<<gemm_grid_hid, 256, 0, stream>>>(h0, wp_s1, agg, wp_n1, b1, invdeg, h1,
                                                      N, HID_F, HID_F, FP_HID);
  hipMemsetAsync(bn_sum, 0, 2 * HID_F * sizeof(float), stream);
  col_stats_kernel<<<(N + 127) / 128, HID_F, 0, stream>>>(h1, bn_sum, bn_sq, N, HID_F);
  bn_finalize_kernel<<<1, HID_F, 0, stream>>>(bn_sum, bn_sq, g1, be1, bn_scale, bn_shift, N, HID_F);
  bn_apply_relu_kernel<<<(int)((totHid + 255) / 256), 256, 0, stream>>>(h1, bn_scale, bn_shift, totHid, HID_F);

  // ---- layer 2: h1 (256) -> logits (40), then log_softmax ----
  hipMemsetAsync(agg, 0, (size_t)N * HID_F * sizeof(float), stream);
  {
    long long tot = (long long)E * (HID_F / 4);
    agg_scatter_kernel<<<(int)((tot + 255) / 256), 256, 0, stream>>>(h1, src, dst, agg, tot, HID_F, 6);
  }
  sage_gemm_kernel<<<gemm_grid_cls, 256, 0, stream>>>(h1, wp_s2, agg, wp_n2, b2, invdeg, logits,
                                                      N, HID_F, NCLS_F, FP_CLS);
  log_softmax_kernel<<<(N + 255) / 256, 256, 0, stream>>>(logits, N, NCLS_F);
}
